// PyramidMPNN_MultiSub_29678224016214
// MI455X (gfx1250) — matrix operand order repaired and checked by reading, not compiled, not run on hardware
//
#include <hip/hip_runtime.h>
#include <hip/hip_bf16.h>
#include <math.h>

// ---------------------------------------------------------------------------
// PyramidMPNN on MI455X (gfx1250): bf16 WMMA (16x16x32, f32 accum) for all
// GEMMs, LDS-staged LSTM recurrence, deterministic per-dst GAT aggregation.
// ---------------------------------------------------------------------------

#define HID   300
#define KP    320          // padded K per 300-wide segment (10 chunks of 32)
#define KC1   10           // k-chunks for one segment
#define KC2   20           // k-chunks for [x|h] concat (640)
#define GW    1200         // LSTM gate width (4*HID)
#define NT_G  75           // 1200 / 16
#define NT_O  19           // ceil(300/16) -> 304 cols
#define NT_A  150          // 2400 / 16 (GAT projection)
#define GSTR  1224         // LDS G row stride in floats (1224 % 64 == 8)
#define MAXE  512          // max edges per dst node held in LDS

typedef __attribute__((ext_vector_type(16))) __bf16 v16bf;
typedef __attribute__((ext_vector_type(8)))  float  v8f;

union FragB16 { v16bf v; uint4 q[2]; };

__device__ __forceinline__ unsigned short f2bf(float f) {
  unsigned int u = __float_as_uint(f);
  u += 0x7FFFu + ((u >> 16) & 1u);          // round-to-nearest-even
  return (unsigned short)(u >> 16);
}
__device__ __forceinline__ float bf2f(unsigned short h) {
  return __uint_as_float(((unsigned int)h) << 16);
}
__device__ __forceinline__ float sigmoidf_(float x) {
  return 1.0f / (1.0f + __expf(-x));
}

// --------------------------- prep kernels ----------------------------------

// f32 [rows][scols] -> bf16 [rows][dcols] zero-padded
__global__ void k_f32_to_bf16_pad(const float* __restrict__ src,
                                  unsigned short* __restrict__ dst,
                                  int rows, int scols, int dcols) {
  long long i = (long long)blockIdx.x * blockDim.x + threadIdx.x;
  long long total = (long long)rows * dcols;
  if (i >= total) return;
  int r = (int)(i / dcols), c = (int)(i % dcols);
  dst[i] = (c < scols) ? f2bf(src[(long long)r * scols + c]) : (unsigned short)0;
}

// Pack B[k][n] = Wseg[n][kk]  (transposed weights, optional 2nd K segment).
// Layout: out[((kc*NT + nt)*32 + lane)*16 + e], k = kc*32+(lane>>4)*8+(e&7)+(e>>3)*16, n = nt*16+(lane&15).
__global__ void k_pack_T(const float* __restrict__ W1, const float* __restrict__ W2,
                         int Nout, int Kin, int NT, int KCtot,
                         unsigned short* __restrict__ out) {
  long long i = (long long)blockIdx.x * blockDim.x + threadIdx.x;
  long long total = (long long)KCtot * NT * 512;
  if (i >= total) return;
  int e    = (int)(i & 15);
  int lane = (int)((i >> 4) & 31);
  long long f = i >> 9;
  int nt = (int)(f % NT);
  int kc = (int)(f / NT);
  int k = kc * 32 + (lane >> 4) * 8 + (e & 7) + (e >> 3) * 16;
  int n = nt * 16 + (lane & 15);
  const float* W = W1; int kk = k;
  if (k >= KP) { W = W2; kk = k - KP; }
  float v = 0.0f;
  if (W != nullptr && kk < Kin && n < Nout) v = W[(long long)n * Kin + kk];
  out[i] = f2bf(v);
}

// Pack B[k][n] = W[k][n]  (non-transposed, GAT projection W is [Kin][Nout]).
__global__ void k_pack_N(const float* __restrict__ W, int Nout, int Kin,
                         int NT, int KCtot, unsigned short* __restrict__ out) {
  long long i = (long long)blockIdx.x * blockDim.x + threadIdx.x;
  long long total = (long long)KCtot * NT * 512;
  if (i >= total) return;
  int e    = (int)(i & 15);
  int lane = (int)((i >> 4) & 31);
  long long f = i >> 9;
  int nt = (int)(f % NT);
  int kc = (int)(f / NT);
  int k = kc * 32 + (lane >> 4) * 8 + (e & 7) + (e >> 3) * 16;
  int n = nt * 16 + (lane & 15);
  float v = 0.0f;
  if (k < Kin && n < Nout) v = W[(long long)k * Nout + n];
  out[i] = f2bf(v);
}

// ------------------------- SAGE-LSTM relation ------------------------------
// One workgroup = 16 dst nodes, 8 wave32s. A-tile [16 x 640] bf16 = [x | h].
__global__ __launch_bounds__(256)
void k_sage_lstm(const unsigned short* __restrict__ xsrc_b,   // [Nsrc][KP] bf16
                 const unsigned short* __restrict__ xdst_b,   // [Ndst][KP] bf16
                 const int* __restrict__ nbr, int D, int Ndst,
                 const unsigned short* __restrict__ Bcat,     // packed 20x75
                 const float* __restrict__ bih, const float* __restrict__ bhh,
                 const unsigned short* __restrict__ Bout,     // packed 20x19
                 const float* __restrict__ bout,
                 const float* __restrict__ prelu_w,
                 float* __restrict__ hacc, int accumulate) {
  __shared__ unsigned short Acat[16 * 640];   // 20 KB
  __shared__ float          G[16 * GSTR];     // 76.5 KB
  __shared__ float          Cst[16 * HID];    // 18.75 KB

  const int tid  = threadIdx.x;
  const int lane = tid & 31;
  const int wave = tid >> 5;
  const int base = blockIdx.x * 16;
  const float pw = prelu_w[0];

  for (int i = tid; i < 16 * 640; i += 256) Acat[i] = 0;   // x, h, pads
  for (int i = tid; i < 16 * HID; i += 256) Cst[i] = 0.0f;
  __syncthreads();

  const int half = lane >> 4;
  const int mrow = lane & 15;

  for (int t = 0; t < D; ++t) {
    // gather neighbor rows into Acat[:, 0:320)
    for (int i = tid; i < 16 * (KP / 2); i += 256) {
      int m = i / (KP / 2), c = i % (KP / 2);
      int row = base + m; if (row >= Ndst) row = Ndst - 1;
      int s = nbr[(long long)row * D + t];
      ((unsigned int*)&Acat[m * 640])[c] =
          ((const unsigned int*)&xsrc_b[(long long)s * KP])[c];
    }
    __syncthreads();

    // G = Acat @ Bcat  (16 x 1200, K = 640)
    for (int nt = wave; nt < NT_G; nt += 8) {
      v8f acc = {0.f, 0.f, 0.f, 0.f, 0.f, 0.f, 0.f, 0.f};
      const unsigned short* arow = &Acat[mrow * 640];
      for (int kc = 0; kc < KC2; ++kc) {
        FragB16 a, b;
        a.q[0] = *(const uint4*)(arow + kc * 32 + half * 8);
        a.q[1] = *(const uint4*)(arow + kc * 32 + 16 + half * 8);
        const unsigned short* bp =
            Bcat + (((long long)(kc * NT_G + nt)) * 32 + lane) * 16;
        b.q[0] = *(const uint4*)(bp);
        b.q[1] = *(const uint4*)(bp + 8);
        acc = __builtin_amdgcn_wmma_f32_16x16x32_bf16(
            false, a.v, false, b.v, (short)0, acc, false, false);
      }
      int n  = nt * 16 + (lane & 15);
      int mb = (lane >> 4) * 8;
#pragma unroll
      for (int i2 = 0; i2 < 8; ++i2) G[(mb + i2) * GSTR + n] = acc[i2];
    }
    __syncthreads();

    // gates: i,f,g,o  (torch order), update c (f32 LDS) and h (bf16 -> A-tile)
    for (int i = tid; i < 16 * HID; i += 256) {
      int m = i / HID, j = i % HID;
      float gi = G[m * GSTR + j]       + bih[j]       + bhh[j];
      float gf = G[m * GSTR + j + 300] + bih[j + 300] + bhh[j + 300];
      float gg = G[m * GSTR + j + 600] + bih[j + 600] + bhh[j + 600];
      float go = G[m * GSTR + j + 900] + bih[j + 900] + bhh[j + 900];
      float c  = sigmoidf_(gf) * Cst[i] + sigmoidf_(gi) * tanhf(gg);
      float h  = sigmoidf_(go) * tanhf(c);
      Cst[i] = c;
      Acat[m * 640 + KP + j] = f2bf(h);
    }
    __syncthreads();
  }

  // Acat[:, 0:320) = x_dst rows; out = prelu([x_dst|h] @ [Wself|Wneigh]^T + b)
  for (int i = tid; i < 16 * (KP / 2); i += 256) {
    int m = i / (KP / 2), c = i % (KP / 2);
    int row = base + m; if (row >= Ndst) row = Ndst - 1;
    ((unsigned int*)&Acat[m * 640])[c] =
        ((const unsigned int*)&xdst_b[(long long)row * KP])[c];
  }
  __syncthreads();

  for (int nt = wave; nt < NT_O; nt += 8) {
    v8f acc = {0.f, 0.f, 0.f, 0.f, 0.f, 0.f, 0.f, 0.f};
    const unsigned short* arow = &Acat[mrow * 640];
    for (int kc = 0; kc < KC2; ++kc) {
      FragB16 a, b;
      a.q[0] = *(const uint4*)(arow + kc * 32 + half * 8);
      a.q[1] = *(const uint4*)(arow + kc * 32 + 16 + half * 8);
      const unsigned short* bp =
          Bout + (((long long)(kc * NT_O + nt)) * 32 + lane) * 16;
      b.q[0] = *(const uint4*)(bp);
      b.q[1] = *(const uint4*)(bp + 8);
      acc = __builtin_amdgcn_wmma_f32_16x16x32_bf16(
          false, a.v, false, b.v, (short)0, acc, false, false);
    }
    int n  = nt * 16 + (lane & 15);
    int mb = (lane >> 4) * 8;
#pragma unroll
    for (int i2 = 0; i2 < 8; ++i2) G[(mb + i2) * GSTR + n] = acc[i2];
  }
  __syncthreads();

  for (int i = tid; i < 16 * HID; i += 256) {
    int m = i / HID, j = i % HID;
    int row = base + m;
    if (row < Ndst) {
      float v = G[m * GSTR + j] + bout[j];
      v = (v >= 0.0f) ? v : pw * v;
      long long o = (long long)row * HID + j;
      if (accumulate) hacc[o] += v; else hacc[o] = v;
    }
  }
}

// --------------------------- generic GEMM ----------------------------------
// C[M x Nreal] = A(bf16,[M][KC*32]) @ Bpacked.  MODE 1: bf16 out; MODE 2:
// f32 out = prelu(x + bias).  grid = (ceil(M/16), ceil(NT/8)), 256 threads.
template <int MODE>
__global__ __launch_bounds__(256)
void k_gemm(const unsigned short* __restrict__ A,
            const unsigned short* __restrict__ Bp,
            int NT, int KC, int Nreal, int M,
            const float* __restrict__ bias, const float* __restrict__ prelu_w,
            float* __restrict__ Cf, unsigned short* __restrict__ Cb) {
  __shared__ unsigned short As[16 * 640];
  const int tid  = threadIdx.x;
  const int lane = tid & 31;
  const int wave = tid >> 5;
  const int base = blockIdx.x * 16;
  const int KPAD = KC * 32;

  for (int i = tid; i < 16 * (KPAD / 2); i += 256) {
    int m = i / (KPAD / 2), c = i % (KPAD / 2);
    int row = base + m; if (row >= M) row = M - 1;
    ((unsigned int*)&As[m * KPAD])[c] =
        ((const unsigned int*)&A[(long long)row * KPAD])[c];
  }
  __syncthreads();

  int nt = blockIdx.y * 8 + wave;
  if (nt >= NT) return;

  const int half = lane >> 4;
  const int mrow = lane & 15;
  v8f acc = {0.f, 0.f, 0.f, 0.f, 0.f, 0.f, 0.f, 0.f};
  for (int kc = 0; kc < KC; ++kc) {
    FragB16 a, b;
    const unsigned short* arow = &As[mrow * KPAD + kc * 32];
    a.q[0] = *(const uint4*)(arow + half * 8);
    a.q[1] = *(const uint4*)(arow + 16 + half * 8);
    const unsigned short* bp = Bp + (((long long)(kc * NT + nt)) * 32 + lane) * 16;
    b.q[0] = *(const uint4*)(bp);
    b.q[1] = *(const uint4*)(bp + 8);
    acc = __builtin_amdgcn_wmma_f32_16x16x32_bf16(
        false, a.v, false, b.v, (short)0, acc, false, false);
  }

  int n  = nt * 16 + (lane & 15);
  int mb = (lane >> 4) * 8;
  if (n < Nreal) {
#pragma unroll
    for (int i = 0; i < 8; ++i) {
      int row = base + mb + i;
      if (row < M) {
        float v = acc[i];
        if (MODE == 2) {
          v += bias[n];
          float pw = prelu_w[0];
          v = (v >= 0.0f) ? v : pw * v;
          Cf[(long long)row * Nreal + n] = v;
        } else {
          Cb[(long long)row * Nreal + n] = f2bf(v);
        }
      }
    }
  }
}

// --------------------------- GAT pieces ------------------------------------

// out[n][h] = sum_d F[n][h*300+d] * avec[h][d]   (wave w handles head w)
__global__ __launch_bounds__(256)
void k_gat_attn(const unsigned short* __restrict__ F,
                const float* __restrict__ avec,
                float* __restrict__ out, int N) {
  int n = blockIdx.x; if (n >= N) return;
  int wave = threadIdx.x >> 5, lane = threadIdx.x & 31;
  const unsigned short* f = F + (long long)n * 2400 + wave * 300;
  const float* a = avec + wave * 300;
  float s = 0.0f;
  for (int j = lane; j < 300; j += 32) s += bf2f(f[j]) * a[j];
  for (int off = 16; off; off >>= 1) s += __shfl_down(s, off, 32);
  if (lane == 0) out[(long long)n * 8 + wave] = s;
}

// One workgroup per dst node: deterministic ordered edge list, edge-softmax,
// weighted aggregation, bias + prelu, accumulate into hm[Nm][2400].
__global__ __launch_bounds__(256)
void k_gat_aggregate(const int* __restrict__ src, const int* __restrict__ dst,
                     int E,
                     const float* __restrict__ el, const float* __restrict__ er,
                     const unsigned short* __restrict__ fs,
                     const float* __restrict__ bias,
                     const float* __restrict__ prelu_w,
                     float* __restrict__ hm, int accumulate) {
  __shared__ int   elist[MAXE];
  __shared__ float alph[8 * MAXE];
  __shared__ float accbuf[2400];
  __shared__ int   wcnt[8];
  __shared__ int   cnt_s;

  const int m = blockIdx.x;
  const int tid = threadIdx.x, lane = tid & 31, wave = tid >> 5;
  if (tid == 0) cnt_s = 0;
  for (int i = tid; i < 2400; i += 256) accbuf[i] = 0.0f;
  __syncthreads();

  // ordered collection of edges with dst == m (ballot + block scan)
  for (int basee = 0; basee < E; basee += 256) {
    int t = basee + tid;
    bool hit = (t < E) && (dst[t] == m);
    unsigned int bmask = (unsigned int)__ballot(hit);
    if (lane == 0) wcnt[wave] = __popc(bmask);
    __syncthreads();
    int woff = 0;
    for (int w2 = 0; w2 < wave; ++w2) woff += wcnt[w2];
    int rank = __popc(bmask & ((1u << lane) - 1u));
    if (hit) {
      int pos = cnt_s + woff + rank;
      if (pos < MAXE) elist[pos] = t;
    }
    __syncthreads();
    if (tid == 0) {
      int tot = 0;
      for (int w2 = 0; w2 < 8; ++w2) tot += wcnt[w2];
      cnt_s += tot;
    }
    __syncthreads();
  }
  int cnt = cnt_s < MAXE ? cnt_s : MAXE;

  // per-head edge softmax (wave w <-> head w)
  {
    float erm = er[(long long)m * 8 + wave];
    float mx = -1e30f;
    for (int i = lane; i < cnt; i += 32) {
      int t = elist[i];
      float e = el[(long long)src[t] * 8 + wave] + erm;
      e = (e >= 0.0f) ? e : 0.2f * e;
      mx = fmaxf(mx, e);
    }
    for (int off = 16; off; off >>= 1) mx = fmaxf(mx, __shfl_down(mx, off, 32));
    mx = __shfl(mx, 0, 32);
    if (cnt == 0) mx = 0.0f;
    float den = 0.0f;
    for (int i = lane; i < cnt; i += 32) {
      int t = elist[i];
      float e = el[(long long)src[t] * 8 + wave] + erm;
      e = (e >= 0.0f) ? e : 0.2f * e;
      float ex = __expf(e - mx);
      alph[wave * MAXE + i] = ex;
      den += ex;
    }
    for (int off = 16; off; off >>= 1) den += __shfl_down(den, off, 32);
    den = __shfl(den, 0, 32);
    den = fmaxf(den, 1e-9f);
    for (int i = lane; i < cnt; i += 32) alph[wave * MAXE + i] /= den;
  }
  __syncthreads();

  // weighted sum over edges in deterministic (ascending edge-id) order
  for (int i = 0; i < cnt; ++i) {
    const unsigned short* frow = fs + (long long)src[elist[i]] * 2400;
    for (int c = tid; c < 2400; c += 256) {
      int h = c / 300;
      accbuf[c] += alph[h * MAXE + i] * bf2f(frow[c]);
    }
  }
  __syncthreads();

  float pw = prelu_w[0];
  for (int c = tid; c < 2400; c += 256) {
    float v = accbuf[c] + bias[c];
    v = (v >= 0.0f) ? v : pw * v;
    long long o = (long long)m * 2400 + c;
    if (accumulate) hm[o] += v; else hm[o] = v;
  }
}

// mean over heads -> bf16 padded [N][KP]
__global__ void k_mean_heads(const float* __restrict__ hmacc,
                             unsigned short* __restrict__ out, int N) {
  long long i = (long long)blockIdx.x * blockDim.x + threadIdx.x;
  long long total = (long long)N * KP;
  if (i >= total) return;
  int n = (int)(i / KP), d = (int)(i % KP);
  float v = 0.0f;
  if (d < HID) {
    for (int h = 0; h < 8; ++h) v += hmacc[(long long)n * 2400 + h * 300 + d];
    v *= 0.125f;
  }
  out[i] = f2bf(v);
}

// --------------------------- host launch -----------------------------------

extern "C" void kernel_launch(void* const* d_in, const int* in_sizes, int n_in,
                              void* d_out, int out_size, void* d_ws, size_t ws_size,
                              hipStream_t stream) {
  const float* x_f32[5] = {(const float*)d_in[0], (const float*)d_in[1],
                           (const float*)d_in[2], (const float*)d_in[3],
                           (const float*)d_in[4]};
  const int* nbr_b  = (const int*)d_in[5];
  const int* nbr_c  = (const int*)d_in[6];
  const int* nbr_d  = (const int*)d_in[7];
  const int* nbr_e  = (const int*)d_in[8];
  const int* nbr_js = (const int*)d_in[9];
  const int* nbr_jf = (const int*)d_in[10];
  const int* nbr_jp = (const int*)d_in[11];

  int Ncnt[5];
  for (int i = 0; i < 5; ++i) Ncnt[i] = in_sizes[i] / HID;
  const int Na = Ncnt[0], Ns = Ncnt[1], Nf = Ncnt[2], Np = Ncnt[3], Nm = Ncnt[4];
  const int E = in_sizes[12];

  // params: depth-first, insertion order of setup_inputs()
  int p = 18;
  const float* sage[7][7];   // order: b, c, d, e, js, jf, jp
  for (int r = 0; r < 7; ++r)
    for (int k = 0; k < 7; ++k) sage[r][k] = (const float*)d_in[p++];
  const float* gat[3][4];    // gis, gif, gip : W, al, ar, bias
  for (int g = 0; g < 3; ++g)
    for (int k = 0; k < 4; ++k) gat[g][k] = (const float*)d_in[p++];
  const float* emb[5][2];    // emb_a, emb_s, emb_f, emb_p, emb_m : W, b
  for (int e5 = 0; e5 < 5; ++e5)
    for (int k = 0; k < 2; ++k) emb[e5][k] = (const float*)d_in[p++];
  const float* prelu = (const float*)d_in[p++];

  // workspace carve-out
  char* ws = (char*)d_ws; size_t off = 0;
  auto carve = [&](size_t bytes) -> void* {
    void* r = ws + off;
    off = (off + bytes + 255) & ~(size_t)255;
    return r;
  };
  unsigned short* xb[5];
  for (int i = 0; i < 5; ++i)
    xb[i] = (unsigned short*)carve((size_t)Ncnt[i] * KP * 2);
  unsigned short* Bcat[7];
  for (int r = 0; r < 7; ++r) Bcat[r] = (unsigned short*)carve((size_t)KC2 * NT_G * 512 * 2);
  unsigned short* Bout[7];
  for (int r = 0; r < 7; ++r) Bout[r] = (unsigned short*)carve((size_t)KC2 * NT_O * 512 * 2);
  unsigned short* Bgat[3];
  for (int g = 0; g < 3; ++g) Bgat[g] = (unsigned short*)carve((size_t)KC1 * NT_A * 512 * 2);
  unsigned short* Bemb[5];
  for (int e5 = 0; e5 < 5; ++e5) Bemb[e5] = (unsigned short*)carve((size_t)KC1 * NT_O * 512 * 2);
  float* hacc_a = (float*)carve((size_t)Na * HID * 4);
  float* hacc_s = (float*)carve((size_t)Ns * HID * 4);
  float* hacc_f = (float*)carve((size_t)Nf * HID * 4);
  float* hacc_p = (float*)carve((size_t)Np * HID * 4);
  float* hmacc  = (float*)carve((size_t)Nm * 2400 * 4);
  unsigned short* fsbuf = (unsigned short*)carve((size_t)Ns * 2400 * 2); // max src count
  unsigned short* fdbuf = (unsigned short*)carve((size_t)Nm * 2400 * 2);
  float* elbuf = (float*)carve((size_t)Ns * 8 * 4);
  float* erbuf = (float*)carve((size_t)Nm * 8 * 4);
  unsigned short* hb  = (unsigned short*)carve((size_t)Na * KP * 2);
  unsigned short* hmb = (unsigned short*)carve((size_t)Nm * KP * 2);
  (void)ws_size; (void)n_in; (void)out_size;

  // 1) activations -> padded bf16
  for (int i = 0; i < 5; ++i) {
    long long total = (long long)Ncnt[i] * KP;
    k_f32_to_bf16_pad<<<(unsigned)((total + 255) / 256), 256, 0, stream>>>(
        x_f32[i], xb[i], Ncnt[i], HID, KP);
  }

  // 2) weight packing
  for (int r = 0; r < 7; ++r) {
    long long t1 = (long long)KC2 * NT_G * 512;
    k_pack_T<<<(unsigned)((t1 + 255) / 256), 256, 0, stream>>>(
        sage[r][0], sage[r][1], GW, HID, NT_G, KC2, Bcat[r]);   // [Wih|Whh]^T
    long long t2 = (long long)KC2 * NT_O * 512;
    k_pack_T<<<(unsigned)((t2 + 255) / 256), 256, 0, stream>>>(
        sage[r][4], sage[r][5], HID, HID, NT_O, KC2, Bout[r]);  // [Wself|Wneigh]^T
  }
  for (int g = 0; g < 3; ++g) {
    long long t3 = (long long)KC1 * NT_A * 512;
    k_pack_N<<<(unsigned)((t3 + 255) / 256), 256, 0, stream>>>(
        gat[g][0], 2400, HID, NT_A, KC1, Bgat[g]);
  }
  for (int e5 = 0; e5 < 5; ++e5) {
    long long t4 = (long long)KC1 * NT_O * 512;
    k_pack_T<<<(unsigned)((t4 + 255) / 256), 256, 0, stream>>>(
        emb[e5][0], (const float*)nullptr, HID, HID, NT_O, KC1, Bemb[e5]);
  }

  // 3) SAGE-LSTM relations (sequential on stream -> race-free accumulation)
  struct Rel { int r; const unsigned short* xs; const unsigned short* xd;
               const int* nbr; int D; int N; float* acc; int add; };
  const Rel rels[7] = {
      {0, xb[0], xb[0], nbr_b,  6, Na, hacc_a, 0},
      {1, xb[1], xb[1], nbr_c,  6, Ns, hacc_s, 0},
      {4, xb[0], xb[1], nbr_js, 4, Ns, hacc_s, 1},
      {2, xb[2], xb[2], nbr_d,  6, Nf, hacc_f, 0},
      {5, xb[1], xb[2], nbr_jf, 4, Nf, hacc_f, 1},
      {3, xb[3], xb[3], nbr_e,  6, Np, hacc_p, 0},
      {6, xb[2], xb[3], nbr_jp, 4, Np, hacc_p, 1},
  };
  for (int i = 0; i < 7; ++i) {
    const Rel& R = rels[i];
    k_sage_lstm<<<(unsigned)((R.N + 15) / 16), 256, 0, stream>>>(
        R.xs, R.xd, R.nbr, R.D, R.N, Bcat[R.r],
        sage[R.r][2], sage[R.r][3], Bout[R.r], sage[R.r][6],
        prelu, R.acc, R.add);
  }

  // 4) GAT relations into hmacc [Nm][2400]
  const unsigned short* gsrc[3] = {xb[1], xb[2], xb[3]};
  const int gN[3] = {Ns, Nf, Np};
  const int* gsi[3] = {(const int*)d_in[12], (const int*)d_in[14], (const int*)d_in[16]};
  const int* gdi[3] = {(const int*)d_in[13], (const int*)d_in[15], (const int*)d_in[17]};
  for (int g = 0; g < 3; ++g) {
    dim3 gr1((unsigned)((gN[g] + 15) / 16), (NT_A + 7) / 8);
    k_gemm<1><<<gr1, 256, 0, stream>>>(gsrc[g], Bgat[g], NT_A, KC1, 2400, gN[g],
                                       (const float*)nullptr, (const float*)nullptr,
                                       (float*)nullptr, fsbuf);
    dim3 gr2((unsigned)((Nm + 15) / 16), (NT_A + 7) / 8);
    k_gemm<1><<<gr2, 256, 0, stream>>>(xb[4], Bgat[g], NT_A, KC1, 2400, Nm,
                                       (const float*)nullptr, (const float*)nullptr,
                                       (float*)nullptr, fdbuf);
    k_gat_attn<<<(unsigned)gN[g], 256, 0, stream>>>(fsbuf, gat[g][1], elbuf, gN[g]);
    k_gat_attn<<<(unsigned)Nm, 256, 0, stream>>>(fdbuf, gat[g][2], erbuf, Nm);
    k_gat_aggregate<<<(unsigned)Nm, 256, 0, stream>>>(
        gsi[g], gdi[g], E, elbuf, erbuf, fsbuf, gat[g][3], prelu, hmacc, g > 0);
  }
  {
    long long total = (long long)Nm * KP;
    k_mean_heads<<<(unsigned)((total + 255) / 256), 256, 0, stream>>>(hmacc, hmb, Nm);
  }

  // 5) final per-type update: prelu(h @ Wemb^T + b) -> d_out
  float* hs[4] = {hacc_a, hacc_s, hacc_f, hacc_p};
  int Nt[4] = {Na, Ns, Nf, Np};
  long long ooff = 0;
  for (int i = 0; i < 4; ++i) {
    long long total = (long long)Nt[i] * KP;
    k_f32_to_bf16_pad<<<(unsigned)((total + 255) / 256), 256, 0, stream>>>(
        hs[i], hb, Nt[i], HID, KP);
    dim3 gu((unsigned)((Nt[i] + 15) / 16), (NT_O + 7) / 8);
    k_gemm<2><<<gu, 256, 0, stream>>>(hb, Bemb[i], NT_O, KC1, HID, Nt[i],
                                      emb[i][1], prelu,
                                      (float*)d_out + ooff, (unsigned short*)nullptr);
    ooff += (long long)Nt[i] * HID;
  }
  dim3 gm((unsigned)((Nm + 15) / 16), (NT_O + 7) / 8);
  k_gemm<2><<<gm, 256, 0, stream>>>(hmb, Bemb[4], NT_O, KC1, HID, Nm,
                                    emb[4][1], prelu,
                                    (float*)d_out + ooff, (unsigned short*)nullptr);
}
